// ConvAE_29661044146819
// MI455X (gfx1250) — matrix operand order repaired and checked by reading, not compiled, not run on hardware
//
#include <hip/hip_runtime.h>
#include <hip/hip_bf16.h>

typedef __attribute__((ext_vector_type(16))) __bf16 v16bf;
typedef __attribute__((ext_vector_type(8)))  float  v8f;

#define BB 128
#define NFC 128

// ---------- bf16 helpers (round-to-nearest-even) ----------
__device__ __forceinline__ unsigned short f2bf(float f) {
    union { float f; unsigned u; } x; x.f = f;
    unsigned r = x.u + 0x7FFFu + ((x.u >> 16) & 1u);
    return (unsigned short)(r >> 16);
}

// low 32 bits of a generic/LDS pointer = LDS byte address for ds/async ops
__device__ __forceinline__ unsigned lds_addr32(const void* p) {
    return (unsigned)(unsigned long long)p;
}

// ---------- weight f32 -> bf16 ----------
__global__ void k_f32_to_bf16(const float* __restrict__ src,
                              unsigned short* __restrict__ dst, int n) {
    int i = blockIdx.x * blockDim.x + threadIdx.x;
    if (i < n) dst[i] = f2bf(src[i]);
}

// ---------- conv1: C=1, 5x5, 64->60, +bias, ReLU, bf16 out ----------
__global__ __launch_bounds__(256)
void k_conv1_relu(const float* __restrict__ X, const float* __restrict__ w1,
                  const float* __restrict__ b1, unsigned short* __restrict__ out) {
    __shared__ float ws[NFC * 25];
    for (int t = threadIdx.x; t < NFC * 25; t += 256) ws[t] = w1[t];
    __syncthreads();
    long long i = (long long)blockIdx.x * 256 + threadIdx.x;
    const long long total = (long long)BB * NFC * 60 * 60;
    if (i >= total) return;
    int ox = (int)(i % 60);
    int oy = (int)((i / 60) % 60);
    int m  = (int)((i / 3600) % NFC);
    int b  = (int)(i / (3600LL * NFC));
    const float* xb = X + (long long)b * 64 * 64;
    const float* wm = ws + m * 25;
    float acc = b1[m];
#pragma unroll
    for (int ky = 0; ky < 5; ++ky) {
        const float* xr = xb + (oy + ky) * 64 + ox;
#pragma unroll
        for (int kx = 0; kx < 5; ++kx) acc += xr[kx] * wm[ky * 5 + kx];
    }
    out[i] = f2bf(acc > 0.f ? acc : 0.f);
}

// ---------- implicit-GEMM 5x5 conv, IC=128, bf16 WMMA, f32 accum ----------
// M=128 out channels, N=B*OH*OW pixels, K=3200. WG = 256 thr = 8 wave32.
// Tile: M=128 x N=128, K-step 32, double-buffered LDS, async weight staging.
// Wave w owns a 2(M)x4(N) block of 16x16 fragments: 12 DS loads per 8 WMMAs.
template<int IH, int IW, int OH, int OW, bool RELU, bool OUT_BF16>
__global__ __launch_bounds__(256)
void k_conv5x5_wmma(const unsigned short* __restrict__ in,   // bf16 [B,128,IH,IW]
                    const unsigned short* __restrict__ wbf,  // bf16 [128,3200]
                    const float* __restrict__ bias,
                    unsigned short* __restrict__ out_bf,
                    float* __restrict__ out_f) {
    constexpr int IC = 128;
    constexpr int K  = IC * 25;           // 3200
    constexpr int NT = 128;
    constexpr int totalN = BB * OH * OW;
    constexpr int RS  = 40;               // padded row stride in halves (80B): 16 rows -> 16 banks
    constexpr int ASz = 128 * RS;         // halves per A buffer
    constexpr int BSz = NT  * RS;         // halves per B buffer

    __shared__ alignas(16) unsigned short At[2 * ASz];
    __shared__ alignas(16) unsigned short Bt[2 * BSz];
    __shared__ alignas(16) int offs[K];   // im2col address offsets per K index

    const int tid    = threadIdx.x;
    const int lane   = tid & 31;
    const int wv     = tid >> 5;          // 0..7
    const int laneLo = lane & 15;
    const int laneHi = lane >> 4;
    const int mg     = wv & 3;            // M group: strips 2mg, 2mg+1
    const int ng     = wv >> 2;           // N group: strips 4ng .. 4ng+3
    const int tileN  = blockIdx.x * NT;

    // Build im2col offset table once: offs[k] = (ic*IH + ky)*IW + kx
    for (int t = tid; t < K; t += 256) {
        const int ic = t / 25, r = t - ic * 25, ky = r / 5, kx = r - ky * 5;
        offs[t] = (ic * IH + ky) * IW + kx;
    }

    // Per-thread staging assignment: thread -> (row 0..127, half 0/1)
    const int sRow  = tid >> 1;
    const int sHalf = tid & 1;
    const int pStage = tileN + sRow;
    const bool pValid = pStage < totalN;
    int pixBase = 0;
    if (pValid) {
        const int sox = pStage % OW;
        const int t   = pStage / OW;
        const int soy = t % OH;
        const int sb  = t / OH;
        pixBase = (sb * IC * IH + soy) * IW + sox;
    }
    __syncthreads();   // offs table ready

    v8f acc[8];        // acc[mi*4+ni]
#pragma unroll
    for (int i = 0; i < 8; ++i)
#pragma unroll
        for (int j = 0; j < 8; ++j) acc[i][j] = 0.f;

    // ---- async weight tile staging: 32B per thread via 2x b128 ----
    auto stageA = [&](int buf, int k0) {
        const unsigned ldsa = lds_addr32(At + buf * ASz + sRow * RS + sHalf * 16);
        const unsigned long long ga =
            (unsigned long long)(const void*)(wbf + sRow * K + k0 + sHalf * 16);
        asm volatile(
            "global_load_async_to_lds_b128 %0, %1, off\n\t"
            "global_load_async_to_lds_b128 %0, %1, off offset:16"
            :: "v"(ldsa), "v"(ga) : "memory");
    };

    // ---- im2col gather staging: 16 halves per thread ----
    auto stageB = [&](int buf, int k0) {
        const int kk = k0 + sHalf * 16;
        union { uint4 q[4]; int s[16]; } of;
        const uint4* op = (const uint4*)(offs + kk);
        of.q[0] = op[0]; of.q[1] = op[1]; of.q[2] = op[2]; of.q[3] = op[3];
        union { uint4 u[2]; unsigned short s[16]; } pk;
#pragma unroll
        for (int j = 0; j < 16; ++j)
            pk.s[j] = pValid ? in[pixBase + of.s[j]] : (unsigned short)0;
        unsigned short* dst = Bt + buf * BSz + sRow * RS + sHalf * 16;
        *(uint4*)dst       = pk.u[0];
        *(uint4*)(dst + 8) = pk.u[1];
    };

    // ---- fragment loads + 8 WMMAs per wave (2 A frags x 4 B frags) ----
    auto compute = [&](int buf) {
        // A 16x32 bf16: lane = row laneLo; elems 0..7 -> K=laneHi*8+e, 8..15 -> 16+laneHi*8+e
        union AFu { uint4 u[2]; v16bf v; } af[2];
#pragma unroll
        for (int mi = 0; mi < 2; ++mi) {
            const unsigned short* ar =
                At + buf * ASz + ((mg * 2 + mi) * 16 + laneLo) * RS;
            af[mi].u[0] = *(const uint4*)(ar + laneHi * 8);
            af[mi].u[1] = *(const uint4*)(ar + 16 + laneHi * 8);
        }
        // B 32x16 bf16: lane = col laneLo; elems 0..7 -> K=laneHi*8+e, 8..15 -> 16+laneHi*8+e
        union BFu { uint4 u[2]; v16bf v; } bf[4];
#pragma unroll
        for (int ni = 0; ni < 4; ++ni) {
            const unsigned short* br =
                Bt + buf * BSz + ((ng * 4 + ni) * 16 + laneLo) * RS;
            bf[ni].u[0] = *(const uint4*)(br + laneHi * 8);
            bf[ni].u[1] = *(const uint4*)(br + 16 + laneHi * 8);
        }
#pragma unroll
        for (int mi = 0; mi < 2; ++mi)
#pragma unroll
            for (int ni = 0; ni < 4; ++ni)
                acc[mi * 4 + ni] = __builtin_amdgcn_wmma_f32_16x16x32_bf16(
                    false, af[mi].v, false, bf[ni].v, (short)0,
                    acc[mi * 4 + ni], false, false);
    };

    // ---- ping-pong main loop ----
    stageA(0, 0);
    stageB(0, 0);
    asm volatile("s_wait_asynccnt 0x0" ::: "memory");
    __syncthreads();
    int p = 0;
    for (int k0 = 0; k0 < K; k0 += 32) {
        if (k0 + 32 < K) {
            stageA(p ^ 1, k0 + 32);
            stageB(p ^ 1, k0 + 32);
        }
        compute(p);
        asm volatile("s_wait_asynccnt 0x0" ::: "memory");
        __syncthreads();
        p ^= 1;
    }

    // ---- epilogue: bias (+ReLU), store ----
#pragma unroll
    for (int mi = 0; mi < 2; ++mi)
#pragma unroll
        for (int ni = 0; ni < 4; ++ni) {
            const int pp = tileN + (ng * 4 + ni) * 16 + laneLo;
            if (pp >= totalN) continue;
            const int ox = pp % OW;
            const int t  = pp / OW;
            const int oy = t % OH;
            const int b  = t / OH;
#pragma unroll
            for (int r = 0; r < 8; ++r) {
                const int m = (mg * 2 + mi) * 16 + laneHi * 8 + r;  // C/D: VGPR r -> M=laneHi*8+r
                float v = acc[mi * 4 + ni][r] + bias[m];
                if (RELU) v = v > 0.f ? v : 0.f;
                const int oidx = ((b * NFC + m) * OH + oy) * OW + ox;
                if (OUT_BF16) out_bf[oidx] = f2bf(v);
                else          out_f[oidx]  = v;
            }
        }
}

// ---------- spatial sparsity: keep per-(b,f) spatial max (ties kept) ----------
__global__ __launch_bounds__(256)
void k_spatial_sparsity(float* __restrict__ h) {
    __shared__ float red[256];
    float* p = h + (long long)blockIdx.x * 2704;   // 52*52
    float m = -3.402823466e38f;
    for (int i = threadIdx.x; i < 2704; i += 256) m = fmaxf(m, p[i]);
    red[threadIdx.x] = m; __syncthreads();
    for (int s = 128; s > 0; s >>= 1) {
        if (threadIdx.x < s) red[threadIdx.x] = fmaxf(red[threadIdx.x], red[threadIdx.x + s]);
        __syncthreads();
    }
    m = red[0];
    for (int i = threadIdx.x; i < 2704; i += 256) {
        float v = p[i];
        p[i] = (v == m) ? v : 0.f;
    }
}

// ---------- strided channel sparsity: max over (f, 4x4 block) ----------
__global__ __launch_bounds__(256)
void k_channel_sparsity(float* __restrict__ h) {
    __shared__ float red[256];
    const int blk = blockIdx.x;                 // b*169 + by*13 + bx
    const int bx = blk % 13;
    const int t0 = blk / 13;
    const int by = t0 % 13;
    const int b  = t0 / 13;
    float m = -3.402823466e38f;
    for (int j = threadIdx.x; j < 2048; j += 256) {
        const int f = j >> 4, dy = (j >> 2) & 3, dx = j & 3;
        m = fmaxf(m, h[((b * NFC + f) * 52 + by * 4 + dy) * 52 + bx * 4 + dx]);
    }
    red[threadIdx.x] = m; __syncthreads();
    for (int s = 128; s > 0; s >>= 1) {
        if (threadIdx.x < s) red[threadIdx.x] = fmaxf(red[threadIdx.x], red[threadIdx.x + s]);
        __syncthreads();
    }
    m = red[0];
    for (int j = threadIdx.x; j < 2048; j += 256) {
        const int f = j >> 4, dy = (j >> 2) & 3, dx = j & 3;
        const int idx = ((b * NFC + f) * 52 + by * 4 + dy) * 52 + bx * 4 + dx;
        const float v = h[idx];
        h[idx] = (v == m) ? v : 0.f;
    }
}

// ---------- deconv: zero accumulator, sparse scatter, sigmoid ----------
__global__ void k_zero(float* p, int n) {
    int i = blockIdx.x * blockDim.x + threadIdx.x;
    if (i < n) p[i] = 0.f;
}

__global__ __launch_bounds__(256)
void k_deconv_scatter(const float* __restrict__ h, const float* __restrict__ wd,
                      float* __restrict__ acc) {
    long long i = (long long)blockIdx.x * 256 + threadIdx.x;
    const long long total = (long long)BB * NFC * 52 * 52;
    if (i >= total) return;
    const float v = h[i];
    if (v == 0.f) return;
    const int x = (int)(i % 52);
    long long t = i / 52;
    const int y = (int)(t % 52); t /= 52;
    const int f = (int)(t % NFC);
    const int b = (int)(t / NFC);
    const float* w = wd + f * 169;              // wd: (NF, 1, 13, 13)
    float* ab = acc + b * 4096;
#pragma unroll 1
    for (int ky = 0; ky < 13; ++ky)
#pragma unroll
        for (int kx = 0; kx < 13; ++kx)
            atomicAdd(&ab[(y + ky) * 64 + (x + kx)], v * w[ky * 13 + kx]);
}

__global__ void k_sigmoid(const float* __restrict__ acc, const float* __restrict__ bd,
                          float* __restrict__ out, int n) {
    int i = blockIdx.x * blockDim.x + threadIdx.x;
    if (i < n) {
        float v = acc[i] + bd[0];
        out[i] = 1.f / (1.f + __expf(-v));
    }
}

extern "C" void kernel_launch(void* const* d_in, const int* in_sizes, int n_in,
                              void* d_out, int out_size, void* d_ws, size_t ws_size,
                              hipStream_t stream) {
    const float* X  = (const float*)d_in[0];
    const float* w1 = (const float*)d_in[1];
    const float* b1 = (const float*)d_in[2];
    const float* w2 = (const float*)d_in[3];
    const float* b2 = (const float*)d_in[4];
    const float* w3 = (const float*)d_in[5];
    const float* b3 = (const float*)d_in[6];
    const float* wd = (const float*)d_in[7];
    const float* bd = (const float*)d_in[8];

    char* ws = (char*)d_ws;
    size_t off = 0;
    auto alloc = [&](size_t bytes) { size_t o = off; off = (off + bytes + 255) & ~(size_t)255; return o; };
    const int KW = NFC * NFC * 25;                                   // 409600
    unsigned short* w2bf = (unsigned short*)(ws + alloc((size_t)KW * 2));
    unsigned short* w3bf = (unsigned short*)(ws + alloc((size_t)KW * 2));
    char*  bufA = ws + alloc((size_t)BB * NFC * 52 * 52 * 4);        // a1(bf16) then h3(f32)
    unsigned short* a2 = (unsigned short*)(ws + alloc((size_t)BB * NFC * 56 * 56 * 2));
    float* acc = (float*)(ws + alloc((size_t)BB * 64 * 64 * 4));

    unsigned short* a1 = (unsigned short*)bufA;
    float* h3 = (float*)bufA;

    k_f32_to_bf16<<<(KW + 255) / 256, 256, 0, stream>>>(w2, w2bf, KW);
    k_f32_to_bf16<<<(KW + 255) / 256, 256, 0, stream>>>(w3, w3bf, KW);

    const long long tot1 = (long long)BB * NFC * 60 * 60;            // 58,982,400
    k_conv1_relu<<<(int)((tot1 + 255) / 256), 256, 0, stream>>>(X, w1, b1, a1);

    const int totN2 = BB * 56 * 56;                                  // 401,408
    k_conv5x5_wmma<60, 60, 56, 56, true, true>
        <<<(totN2 + 127) / 128, 256, 0, stream>>>(a1, w2bf, b2, a2, nullptr);

    const int totN3 = BB * 52 * 52;                                  // 346,112
    k_conv5x5_wmma<56, 56, 52, 52, false, false>
        <<<(totN3 + 127) / 128, 256, 0, stream>>>(a2, w3bf, b3, nullptr, h3);

    k_spatial_sparsity<<<BB * NFC, 256, 0, stream>>>(h3);
    k_channel_sparsity<<<BB * 169, 256, 0, stream>>>(h3);

    const int nOut = BB * 64 * 64;                                   // 524,288
    k_zero<<<(nOut + 255) / 256, 256, 0, stream>>>(acc, nOut);
    const long long totH = (long long)BB * NFC * 52 * 52;
    k_deconv_scatter<<<(int)((totH + 255) / 256), 256, 0, stream>>>(h3, wd, acc);
    k_sigmoid<<<(nOut + 255) / 256, 256, 0, stream>>>(acc, bd, (float*)d_out, nOut);
}
